// SPR_GNN_88648124990730
// MI455X (gfx1250) — matrix operand order repaired and checked by reading, not compiled, not run on hardware
//
#include <hip/hip_runtime.h>
#include <hip/hip_bf16.h>

#define N_NODES   50000
#define N_EDGES   800000
#define EMBED     64
#define HIDDEN    64
#define N_CLASSES 10
#define N_GRAPHS  512

typedef __attribute__((ext_vector_type(16))) _Float16 v16h;
typedef __attribute__((ext_vector_type(8)))  _Float16 v8h;
typedef __attribute__((ext_vector_type(8)))  float    v8f;

// ---------------- small utility kernels ----------------

// Wt[n*64+k] = (f16) W[k*64+n]  — transposed f16 weights so B fragments are contiguous
__global__ void k_weights_prep(const float* __restrict__ W1, const float* __restrict__ W2,
                               _Float16* __restrict__ W1t, _Float16* __restrict__ W2t) {
    int i = blockIdx.x * blockDim.x + threadIdx.x;
    if (i >= EMBED * HIDDEN) return;
    int n = i >> 6, k = i & 63;
    W1t[i] = (_Float16)W1[k * HIDDEN + n];
    W2t[i] = (_Float16)W2[k * HIDDEN + n];
}

__global__ void k_fill(float* __restrict__ p, float v, int n) {
    int i = blockIdx.x * blockDim.x + threadIdx.x;
    if (i < n) p[i] = v;
}

__global__ void k_deg_accum(const int* __restrict__ dst, float* __restrict__ deg, int ne) {
    int e = blockIdx.x * blockDim.x + threadIdx.x;
    if (e < ne) atomicAdd(&deg[dst[e]], 1.0f);
}

__global__ void k_rsqrt(const float* __restrict__ deg, float* __restrict__ dis, int n) {
    int i = blockIdx.x * blockDim.x + threadIdx.x;
    if (i < n) dis[i] = rsqrtf(deg[i]);
}

// x[node, :] = emb[tokens[node], :]  — float4 per thread (16 threads / node)
__global__ void k_gather_embed(const int* __restrict__ tokens, const float* __restrict__ emb,
                               float* __restrict__ x, int n_nodes) {
    int t = blockIdx.x * blockDim.x + threadIdx.x;
    int total = n_nodes * (EMBED / 4);
    if (t >= total) return;
    int node = t >> 4;           // EMBED/4 == 16
    int c4   = t & 15;
    const float4* src = (const float4*)(emb + (size_t)tokens[node] * EMBED);
    float4*       dstp = (float4*)(x + (size_t)node * EMBED);
    dstp[c4] = src[c4];
}

// ---------------- WMMA GEMM: h[N,64] = x[N,64] @ W[64,64] ----------------
// Wt is the f16 weight matrix stored transposed: Wt[n][k].
// One wave per 16x64 row strip: A fragments loaded once (both K=32 halves),
// reused across 4 column tiles -> 8 v_wmma per wave; each B fragment is
// two runs of 8 contiguous halfs of Wt, 16 K apart -> two b128 loads.

__global__ void k_gemm_wmma(const float* __restrict__ x, const _Float16* __restrict__ Wt,
                            float* __restrict__ h, int n_rows) {
    const int lane  = threadIdx.x & 31;
    const int wave  = threadIdx.x >> 5;
    const int strip = blockIdx.x * (blockDim.x >> 5) + wave;
    const int nStrips = n_rows >> 4;               // exact: 50000 = 3125*16
    if (strip >= nStrips) return;                  // wave-uniform: EXEC stays all-1

    const int  m0  = strip * 16;
    const bool hi  = lane >= 16;
    const int  l15 = lane & 15;
    const float* xr = x + (size_t)(m0 + l15) * EMBED;

    // A fragments (16x32 f16 layout): lo lanes K={kb..kb+7, kb+16..kb+23},
    // hi lanes K={kb+8..kb+15, kb+24..kb+31}; one per K=32 half of EMBED=64.
    v16h a0, a1;
    {
        const int kA0 = hi ? 8 : 0;
        const int kA1 = hi ? 24 : 16;
#pragma unroll
        for (int i = 0; i < 8; ++i) a0[i]     = (_Float16)xr[kA0 + i];
#pragma unroll
        for (int i = 0; i < 8; ++i) a0[8 + i] = (_Float16)xr[kA1 + i];
#pragma unroll
        for (int i = 0; i < 8; ++i) a1[i]     = (_Float16)xr[32 + kA0 + i];
#pragma unroll
        for (int i = 0; i < 8; ++i) a1[8 + i] = (_Float16)xr[32 + kA1 + i];
    }

    const int mbase = m0 + (hi ? 8 : 0);
    const int kB    = hi ? 8 : 0;                  // B: lo lanes K={0..7,16..23}, hi lanes K={8..15,24..31}

#pragma unroll
    for (int nt = 0; nt < HIDDEN / 16; ++nt) {
        const int c = nt * 16 + l15;
        const v8h* wc = (const v8h*)(Wt + (size_t)c * EMBED);   // column c of W, contiguous in K

        union { v16h v; v8h p[2]; } b0u, b1u;
        b0u.p[0] = wc[(kB >> 3)];                  // K = kB .. kB+7
        b0u.p[1] = wc[(kB >> 3) + 2];              // K = kB+16 .. kB+23
        b1u.p[0] = wc[(kB >> 3) + 4];              // K = 32+kB .. 32+kB+7
        b1u.p[1] = wc[(kB >> 3) + 6];              // K = 32+kB+16 .. 32+kB+23

        v8f acc = {};
        acc = __builtin_amdgcn_wmma_f32_16x16x32_f16(false, a0, false, b0u.v,
                                                     (short)0, acc, false, false);
        acc = __builtin_amdgcn_wmma_f32_16x16x32_f16(false, a1, false, b1u.v,
                                                     (short)0, acc, false, false);

        // C/D layout: VGPR j -> M = mbase + j, N = c
#pragma unroll
        for (int j = 0; j < 8; ++j)
            h[(size_t)(mbase + j) * HIDDEN + c] = acc[j];
    }
}

// ---------------- edge scatter: agg[dst] += h[src] * dis[src]*dis[dst] ----------------
// One wave per edge; each lane handles 2 columns (coalesced 256B row).

__global__ void k_scatter(const int* __restrict__ src, const int* __restrict__ dst,
                          const float* __restrict__ dis, const float* __restrict__ h,
                          float* __restrict__ agg, int ne) {
    const int lane = threadIdx.x & 31;
    const int e = blockIdx.x * (blockDim.x >> 5) + (threadIdx.x >> 5);
    if (e >= ne) return;
    const int s = src[e], d = dst[e];
    const float coef = dis[s] * dis[d];
    const float2 hv = *(const float2*)(h + (size_t)s * HIDDEN + lane * 2);
    float* ap = agg + (size_t)d * HIDDEN + lane * 2;
    atomicAdd(ap,     hv.x * coef);
    atomicAdd(ap + 1, hv.y * coef);
}

// out = relu(agg + h*dis^2 + bias), written over x for the next layer
__global__ void k_self_relu(const float* __restrict__ agg, const float* __restrict__ h,
                            const float* __restrict__ dis, const float* __restrict__ bias,
                            float* __restrict__ out, int n_nodes) {
    int i = blockIdx.x * blockDim.x + threadIdx.x;
    int total = n_nodes * HIDDEN;
    if (i >= total) return;
    int row = i >> 6, col = i & 63;
    float dd = dis[row];
    float v = agg[i] + h[i] * dd * dd + bias[col];
    out[i] = v > 0.0f ? v : 0.0f;
}

// ---------------- pooling + head ----------------

__global__ void k_pool_accum(const int* __restrict__ batch, const float* __restrict__ x,
                             float* __restrict__ sums, float* __restrict__ cnts, int n_nodes) {
    const int lane = threadIdx.x & 31;
    const int node = blockIdx.x * (blockDim.x >> 5) + (threadIdx.x >> 5);
    if (node >= n_nodes) return;
    const int g = batch[node];
    const float2 xv = *(const float2*)(x + (size_t)node * HIDDEN + lane * 2);
    float* sp = sums + (size_t)g * HIDDEN + lane * 2;
    atomicAdd(sp,     xv.x);
    atomicAdd(sp + 1, xv.y);
    if (lane == 0) atomicAdd(&cnts[g], 1.0f);
}

__global__ void k_head(const float* __restrict__ sums, const float* __restrict__ cnts,
                       const float* __restrict__ linW, const float* __restrict__ linB,
                       float* __restrict__ out) {
    int i = blockIdx.x * blockDim.x + threadIdx.x;
    if (i >= N_GRAPHS * N_CLASSES) return;
    int g = i / N_CLASSES, c = i % N_CLASSES;
    float inv = 1.0f / fmaxf(cnts[g], 1.0f);
    float acc = linB[c];
    const float* sg = sums + (size_t)g * HIDDEN;
#pragma unroll
    for (int j = 0; j < HIDDEN; ++j)
        acc += sg[j] * inv * linW[j * N_CLASSES + c];
    out[i] = acc;
}

// ---------------- launcher ----------------

extern "C" void kernel_launch(void* const* d_in, const int* in_sizes, int n_in,
                              void* d_out, int out_size, void* d_ws, size_t ws_size,
                              hipStream_t stream) {
    const int*   tokens = (const int*)d_in[0];
    const int*   eidx   = (const int*)d_in[1];
    const int*   batch  = (const int*)d_in[2];
    const float* emb    = (const float*)d_in[3];
    const float* W1     = (const float*)d_in[4];
    const float* b1     = (const float*)d_in[5];
    const float* W2     = (const float*)d_in[6];
    const float* b2     = (const float*)d_in[7];
    const float* linW   = (const float*)d_in[8];
    const float* linB   = (const float*)d_in[9];
    float* out = (float*)d_out;

    const int* e_src = eidx;
    const int* e_dst = eidx + N_EDGES;

    // workspace layout (floats)
    float* ws   = (float*)d_ws;
    float* x    = ws;                         // 3,200,000
    float* h    = x   + (size_t)N_NODES * HIDDEN;
    float* agg  = h   + (size_t)N_NODES * HIDDEN;
    float* deg  = agg + (size_t)N_NODES * HIDDEN;
    float* dis  = deg + N_NODES;
    float* sums = dis + N_NODES;              // 512*64
    float* cnts = sums + N_GRAPHS * HIDDEN;   // 512
    _Float16* W1t = (_Float16*)(cnts + N_GRAPHS);   // transposed f16 weights
    _Float16* W2t = W1t + EMBED * HIDDEN;

    const int T = 256;
    const int featTot  = N_NODES * HIDDEN;                 // 3.2M
    const int featBlks = (featTot + T - 1) / T;
    const int nodeBlks = (N_NODES + T - 1) / T;
    const int edgeBlks = (N_EDGES + T - 1) / T;
    const int waveEdgeBlks = (N_EDGES + 7) / 8;            // 8 waves/block, 1 edge/wave
    const int waveNodeBlks = (N_NODES + 7) / 8;
    const int gemmStrips = N_NODES / 16;                   // 3125 (exact)
    const int gemmBlks   = (gemmStrips + 7) / 8;           // 8 waves/block, 1 strip/wave

    // weights -> f16, transposed (both layers at once)
    k_weights_prep<<<(EMBED * HIDDEN + T - 1) / T, T, 0, stream>>>(W1, W2, W1t, W2t);

    // degree (self-loop included) and D^-1/2
    k_fill<<<nodeBlks, T, 0, stream>>>(deg, 1.0f, N_NODES);
    k_deg_accum<<<edgeBlks, T, 0, stream>>>(e_dst, deg, N_EDGES);
    k_rsqrt<<<nodeBlks, T, 0, stream>>>(deg, dis, N_NODES);

    // x = emb[tokens]
    k_gather_embed<<<(N_NODES * 16 + T - 1) / T, T, 0, stream>>>(tokens, emb, x, N_NODES);

    // ---- GCN layer 1 ----
    k_gemm_wmma<<<gemmBlks, T, 0, stream>>>(x, W1t, h, N_NODES);
    k_fill<<<featBlks, T, 0, stream>>>(agg, 0.0f, featTot);
    k_scatter<<<waveEdgeBlks, T, 0, stream>>>(e_src, e_dst, dis, h, agg, N_EDGES);
    k_self_relu<<<featBlks, T, 0, stream>>>(agg, h, dis, b1, x, N_NODES);

    // ---- GCN layer 2 ----
    k_gemm_wmma<<<gemmBlks, T, 0, stream>>>(x, W2t, h, N_NODES);
    k_fill<<<featBlks, T, 0, stream>>>(agg, 0.0f, featTot);
    k_scatter<<<waveEdgeBlks, T, 0, stream>>>(e_src, e_dst, dis, h, agg, N_EDGES);
    k_self_relu<<<featBlks, T, 0, stream>>>(agg, h, dis, b2, x, N_NODES);

    // ---- global mean pool + linear head ----
    k_fill<<<(N_GRAPHS * HIDDEN + T - 1) / T, T, 0, stream>>>(sums, 0.0f, N_GRAPHS * HIDDEN);
    k_fill<<<(N_GRAPHS + T - 1) / T, T, 0, stream>>>(cnts, 0.0f, N_GRAPHS);
    k_pool_accum<<<waveNodeBlks, T, 0, stream>>>(batch, x, sums, cnts, N_NODES);
    k_head<<<(N_GRAPHS * N_CLASSES + T - 1) / T, T, 0, stream>>>(sums, cnts, linW, linB, out);
}
